// MoELayer_2284922601834
// MI455X (gfx1250) — compile-verified, matrix operand order checked by reading, not run on hardware
//
#include <hip/hip_runtime.h>
#include <hip/hip_bf16.h>
#include <cstddef>

typedef __attribute__((ext_vector_type(16))) _Float16 v16h;
typedef __attribute__((ext_vector_type(4)))  _Float16 v4h;
typedef __attribute__((ext_vector_type(8)))  float    v8f;

#define NTOK 2048   // B*S
#define DIM  1024   // D
#define HID  2048   // H
#define NEXP 8      // E
#define HC2  256    // H chunk for the M=32 kernel
#define HCHUNK 512  // H chunk for the fallback M=16 kernel

#define OUT_N   (NTOK * DIM)     // 2,097,152
#define AUX_OFF (OUT_N)
#define GS_OFF  (OUT_N + 1)

// ---- workspace layout ----------------------------------------------------
static constexpr size_t WS_SMALL = (size_t)(8 + NEXP * NTOK) * sizeof(int)
                                 + (size_t)2 * NTOK * sizeof(float);
static constexpr size_t XH_OFF  = (WS_SMALL + 255) & ~(size_t)255;
static constexpr size_t W1T_OFF = XH_OFF  + (size_t)NTOK * DIM * 2;
static constexpr size_t W2T_OFF = W1T_OFF + (size_t)NEXP * DIM * HID * 2;
static constexpr size_t W3T_OFF = W2T_OFF + (size_t)NEXP * DIM * HID * 2;
static constexpr size_t WS_NEED = W3T_OFF + (size_t)NEXP * HID * DIM * 2;

// ---------------- kernel 0: zero output accumulator + expert counters ----
__global__ __launch_bounds__(256)
void moe_zero(float* __restrict__ out, int* __restrict__ counts) {
  int i = blockIdx.x * 256 + threadIdx.x;
  if (i < OUT_N) out[i] = 0.0f;
  if (i < NEXP)  counts[i] = 0;
}

// ---------------- prep: x fp32 -> f16 (vectorized) ------------------------
__global__ __launch_bounds__(256)
void moe_cvt_x(const float* __restrict__ x, _Float16* __restrict__ xh) {
  const int i = (blockIdx.x * 256 + threadIdx.x) * 4;
  const float4 v = *(const float4*)(x + i);
  v4h o = { (_Float16)v.x, (_Float16)v.y, (_Float16)v.z, (_Float16)v.w };
  *(v4h*)(xh + i) = o;
}

// ---------------- prep: weights fp32 row-major -> f16 WMMA B-panels -------
template<int K, int N>
__global__ __launch_bounds__(256)
void moe_cvt_panels(const float* __restrict__ src, _Float16* __restrict__ dst) {
  constexpr int KC = K / 32;
  constexpr int NT = N / 16;
  const size_t o = (size_t)blockIdx.x * 256 + threadIdx.x;
  const int j    = (int)(o & 15);
  const int lane = (int)((o >> 4) & 31);
  size_t r = o >> 9;
  const int kc = (int)(r & (KC - 1)); r /= KC;
  const int t  = (int)(r & (NT - 1)); r /= NT;
  const int e  = (int)r;
  const int ks = (lane >> 4) << 3;
  const int k  = kc * 32 + ((j < 8) ? (ks + j) : (ks + j + 8));
  const int n  = t * 16 + (lane & 15);
  dst[o] = (_Float16)src[((size_t)e * K + k) * N + n];
}

// ---------------- kernel 1: gating, one wave32 per token ------------------
__global__ __launch_bounds__(256)
void moe_gate(const float* __restrict__ x, const float* __restrict__ Wg,
              float* __restrict__ gs, float* __restrict__ tokw,
              float* __restrict__ lse2, int* __restrict__ counts,
              int* __restrict__ lists) {
  const int wid  = threadIdx.x >> 5;
  const int lane = threadIdx.x & 31;
  const int t    = blockIdx.x * 8 + wid;

  float acc[NEXP];
#pragma unroll
  for (int e = 0; e < NEXP; ++e) acc[e] = 0.0f;

  const float* xr = x + (size_t)t * DIM;
  for (int d = lane; d < DIM; d += 32) {
    const float xv = xr[d];
    const float* wr = Wg + (size_t)d * NEXP;
#pragma unroll
    for (int e = 0; e < NEXP; ++e) acc[e] += xv * wr[e];
  }
#pragma unroll
  for (int off = 16; off >= 1; off >>= 1) {
#pragma unroll
    for (int e = 0; e < NEXP; ++e) acc[e] += __shfl_xor(acc[e], off, 32);
  }

  if (lane == 0) {
    float mx = acc[0];
#pragma unroll
    for (int e = 1; e < NEXP; ++e) mx = fmaxf(mx, acc[e]);
    float p[NEXP], s = 0.0f;
#pragma unroll
    for (int e = 0; e < NEXP; ++e) { p[e] = __expf(acc[e] - mx); s += p[e]; }
    const float inv = 1.0f / s;
    const float lse = mx + __logf(s);
    lse2[t] = lse * lse;

    int   i1 = -1, i2 = -1;
    float s1 = -1e30f, s2 = -1e30f;
#pragma unroll
    for (int e = 0; e < NEXP; ++e) {
      const float sc = p[e] * inv;
      gs[(size_t)t * NEXP + e] = sc;
      if (sc > s1)      { s2 = s1; i2 = i1; s1 = sc; i1 = e; }
      else if (sc > s2) { s2 = sc; i2 = e; }
    }
    tokw[t] = s1 + s2;
    int p1 = atomicAdd(&counts[i1], 1); lists[i1 * NTOK + p1] = t;
    int p2 = atomicAdd(&counts[i2], 1); lists[i2 * NTOK + p2] = t;
  }
}

// ---------------- kernel 2: aux loss (deterministic tree reduce) ----------
__global__ __launch_bounds__(256)
void moe_aux(const float* __restrict__ gs, const float* __restrict__ lse2,
             float* __restrict__ aux) {
  __shared__ float red[256 * NEXP];
  __shared__ float redl[256];
  const int tid = threadIdx.x;
  float a[NEXP];
#pragma unroll
  for (int e = 0; e < NEXP; ++e) a[e] = 0.0f;
  float al = 0.0f;
  for (int t = tid; t < NTOK; t += 256) {
#pragma unroll
    for (int e = 0; e < NEXP; ++e) a[e] += gs[(size_t)t * NEXP + e];
    al += lse2[t];
  }
#pragma unroll
  for (int e = 0; e < NEXP; ++e) red[tid * NEXP + e] = a[e];
  redl[tid] = al;
  __syncthreads();
  for (int s = 128; s > 0; s >>= 1) {
    if (tid < s) {
#pragma unroll
      for (int e = 0; e < NEXP; ++e) red[tid * NEXP + e] += red[(tid + s) * NEXP + e];
      redl[tid] += redl[tid + s];
    }
    __syncthreads();
  }
  if (tid == 0) {
    float lb = 0.0f;
#pragma unroll
    for (int e = 0; e < NEXP; ++e) {
      const float u = red[e] * (1.0f / NTOK);
      lb -= u * __logf(u + 1e-9f);
    }
    aux[0] = lb + (redl[0] * (1.0f / NTOK)) * 1e-3f;
  }
}

// ---------------- kernel 3a: expert SwiGLU, M=32 tokens, tiled f16 --------
// grid: (NTOK/32 tiles, NEXP); block: 256 threads = 8 wave32.
// Each B panel (2x global_load_b128/lane) is reused by two A row-tiles.
__global__ __launch_bounds__(256)
void moe_expert32(const _Float16* __restrict__ xh,
                  const v16h* __restrict__ W1p, const v16h* __restrict__ W2p,
                  const v16h* __restrict__ W3p,
                  const float* __restrict__ b1, const float* __restrict__ b2,
                  const float* __restrict__ b3,
                  const int* __restrict__ counts, const int* __restrict__ lists,
                  const float* __restrict__ tokw, float* __restrict__ out) {
  const int e   = blockIdx.y;
  const int cnt = counts[e];
  const int m0  = blockIdx.x * 32;
  if (m0 >= cnt) return;

  __shared__ alignas(16) _Float16 xs[32 * DIM];   // 64 KB
  __shared__ alignas(16) _Float16 acts[32 * HC2]; // 16 KB
  __shared__ int   ts[32];
  __shared__ float wt[32];

  const int tid  = threadIdx.x;
  const int wid  = tid >> 5;
  const int lane = tid & 31;
  const int nl   = lane & 15;
  const int ks   = (lane >> 4) << 3;
  const int mo   = (lane >> 4) << 3;

  if (tid < 32) {
    const int idx = m0 + tid;
    const int tok = (idx < cnt) ? lists[e * NTOK + idx] : -1;
    ts[tid] = tok;
    wt[tid] = (tok >= 0) ? tokw[tok] : 0.0f;
  }
  __syncthreads();

  for (int i = tid; i < 32 * DIM / 8; i += 256) {   // 16B copies
    const int m = i >> 7, c = (i & 127) * 8;
    int tok = ts[m]; if (tok < 0) tok = 0;
    *(uint4*)(xs + m * DIM + c) = *(const uint4*)(xh + (size_t)tok * DIM + c);
  }
  __syncthreads();

  v8f oacc[16];                       // [ot][row-tile], 128 VGPRs persistent
#pragma unroll
  for (int i = 0; i < 16; ++i)
#pragma unroll
    for (int r = 0; r < 8; ++r) oacc[i][r] = 0.0f;

  for (int hc = 0; hc < HID / HC2; ++hc) {
    // ---- phase 1: acts[32][HC2] = silu(x*W1+b1) * (x*W2+b2) ----
#pragma unroll 1
    for (int nt = 0; nt < 2; ++nt) {
      const int hloc = (wid * 2 + nt) * 16;
      const int hb   = hc * HC2 + hloc;
      v8f c1a, c1b, c2a, c2b;
#pragma unroll
      for (int r = 0; r < 8; ++r) { c1a[r] = 0.0f; c1b[r] = 0.0f;
                                    c2a[r] = 0.0f; c2b[r] = 0.0f; }
      for (int kb = 0; kb < DIM; kb += 32) {
        v16h a0, a1, w1v, w2v;
#pragma unroll
        for (int j = 0; j < 16; ++j) {
          const int k = (j < 8) ? (ks + j) : (ks + j + 8);
          a0[j] = xs[nl * DIM + kb + k];
          a1[j] = xs[(16 + nl) * DIM + kb + k];
        }
        const size_t pan = ((size_t)(e * (HID / 16) + (hb >> 4)) * (DIM / 32)
                            + (kb >> 5)) * 32 + lane;
        w1v = W1p[pan];
        w2v = W2p[pan];
        c1a = __builtin_amdgcn_wmma_f32_16x16x32_f16(false, a0, false, w1v,
                                                     (short)0, c1a, false, false);
        c1b = __builtin_amdgcn_wmma_f32_16x16x32_f16(false, a1, false, w1v,
                                                     (short)0, c1b, false, false);
        c2a = __builtin_amdgcn_wmma_f32_16x16x32_f16(false, a0, false, w2v,
                                                     (short)0, c2a, false, false);
        c2b = __builtin_amdgcn_wmma_f32_16x16x32_f16(false, a1, false, w2v,
                                                     (short)0, c2b, false, false);
      }
      const float bb1 = b1[e * HID + hb + nl];
      const float bb2 = b2[e * HID + hb + nl];
#pragma unroll
      for (int r = 0; r < 8; ++r) {
        float s1v = c1a[r] + bb1;
        float s2v = c2a[r] + bb2;
        float sg  = __builtin_amdgcn_rcpf(1.0f + __expf(-s1v));
        acts[(r + mo) * HC2 + hloc + nl] = (_Float16)(s1v * sg * s2v);
        s1v = c1b[r] + bb1;
        s2v = c2b[r] + bb2;
        sg  = __builtin_amdgcn_rcpf(1.0f + __expf(-s1v));
        acts[(16 + r + mo) * HC2 + hloc + nl] = (_Float16)(s1v * sg * s2v);
      }
    }
    __syncthreads();

    // ---- phase 2: out tiles += acts_chunk * W3_chunk ----
#pragma unroll 1
    for (int ot = 0; ot < 8; ++ot) {
      const int ncb = wid * 128 + ot * 16;
      for (int kb = 0; kb < HC2; kb += 32) {
        v16h a0, a1, w3v;
#pragma unroll
        for (int j = 0; j < 16; ++j) {
          const int k = (j < 8) ? (ks + j) : (ks + j + 8);
          a0[j] = acts[nl * HC2 + kb + k];
          a1[j] = acts[(16 + nl) * HC2 + kb + k];
        }
        const size_t pan = ((size_t)(e * (DIM / 16) + (ncb >> 4)) * (HID / 32)
                            + ((hc * HC2 + kb) >> 5)) * 32 + lane;
        w3v = W3p[pan];
        oacc[ot * 2 + 0] = __builtin_amdgcn_wmma_f32_16x16x32_f16(
            false, a0, false, w3v, (short)0, oacc[ot * 2 + 0], false, false);
        oacc[ot * 2 + 1] = __builtin_amdgcn_wmma_f32_16x16x32_f16(
            false, a1, false, w3v, (short)0, oacc[ot * 2 + 1], false, false);
      }
    }
    __syncthreads();
  }

  // ---- write back: out[tok] += w * (y + b3) (exactly 2 adds/element) ----
#pragma unroll 1
  for (int ot = 0; ot < 8; ++ot) {
    const int col = wid * 128 + ot * 16 + nl;
    const float b3v = b3[e * DIM + col];
#pragma unroll
    for (int rt = 0; rt < 2; ++rt) {
#pragma unroll
      for (int r = 0; r < 8; ++r) {
        const int tok = ts[rt * 16 + r + mo];
        if (tok >= 0)
          atomicAdd(&out[(size_t)tok * DIM + col],
                    wt[rt * 16 + r + mo] * (oacc[ot * 2 + rt][r] + b3v));
      }
    }
  }
}

// ---------------- kernel 3b: fallback M=16, fp32 weights in-loop ----------
__global__ __launch_bounds__(256)
void moe_expert16(const float* __restrict__ x,
                  const float* __restrict__ W1, const float* __restrict__ W2,
                  const float* __restrict__ W3,
                  const float* __restrict__ b1, const float* __restrict__ b2,
                  const float* __restrict__ b3,
                  const int* __restrict__ counts, const int* __restrict__ lists,
                  const float* __restrict__ tokw, float* __restrict__ out) {
  const int e   = blockIdx.y;
  const int cnt = counts[e];
  const int m0  = blockIdx.x * 16;
  if (m0 >= cnt) return;

  __shared__ alignas(16) _Float16 xs[16 * DIM];
  __shared__ alignas(16) _Float16 acts[16 * HCHUNK];
  __shared__ int   ts[16];
  __shared__ float wt[16];

  const int tid  = threadIdx.x;
  const int wid  = tid >> 5;
  const int lane = tid & 31;
  const int nl   = lane & 15;
  const int ks   = (lane >> 4) << 3;
  const int mo   = (lane >> 4) << 3;

  if (tid < 16) {
    const int idx = m0 + tid;
    const int tok = (idx < cnt) ? lists[e * NTOK + idx] : -1;
    ts[tid] = tok;
    wt[tid] = (tok >= 0) ? tokw[tok] : 0.0f;
  }
  __syncthreads();
  for (int i = tid; i < 16 * DIM; i += 256) {
    const int m = i >> 10, d = i & (DIM - 1);
    int tok = ts[m]; if (tok < 0) tok = 0;
    xs[i] = (_Float16)x[(size_t)tok * DIM + d];
  }
  __syncthreads();

  v8f oacc[8];
#pragma unroll
  for (int i = 0; i < 8; ++i)
#pragma unroll
    for (int r = 0; r < 8; ++r) oacc[i][r] = 0.0f;

  for (int hc = 0; hc < HID / HCHUNK; ++hc) {
#pragma unroll 1
    for (int nt = 0; nt < 4; ++nt) {
      const int hloc = wid * 64 + nt * 16;
      const int hb   = hc * HCHUNK + hloc;
      v8f c1, c2;
#pragma unroll
      for (int r = 0; r < 8; ++r) { c1[r] = 0.0f; c2[r] = 0.0f; }
      for (int kb = 0; kb < DIM; kb += 32) {
        v16h av, w1v, w2v;
        const size_t wb = ((size_t)e * DIM + kb) * HID + hb + nl;
#pragma unroll
        for (int j = 0; j < 16; ++j) {
          const int k = (j < 8) ? (ks + j) : (ks + j + 8);
          av[j]  = xs[nl * DIM + kb + k];
          w1v[j] = (_Float16)W1[wb + (size_t)k * HID];
          w2v[j] = (_Float16)W2[wb + (size_t)k * HID];
        }
        c1 = __builtin_amdgcn_wmma_f32_16x16x32_f16(false, av, false, w1v,
                                                    (short)0, c1, false, false);
        c2 = __builtin_amdgcn_wmma_f32_16x16x32_f16(false, av, false, w2v,
                                                    (short)0, c2, false, false);
      }
      const float bb1 = b1[e * HID + hb + nl];
      const float bb2 = b2[e * HID + hb + nl];
#pragma unroll
      for (int r = 0; r < 8; ++r) {
        const float s1v = c1[r] + bb1;
        const float s2v = c2[r] + bb2;
        const float sg  = __builtin_amdgcn_rcpf(1.0f + __expf(-s1v));
        acts[(r + mo) * HCHUNK + hloc + nl] = (_Float16)(s1v * sg * s2v);
      }
    }
    __syncthreads();
#pragma unroll 1
    for (int ot = 0; ot < 8; ++ot) {
      const int ncb = wid * 128 + ot * 16;
      for (int kb = 0; kb < HCHUNK; kb += 32) {
        v16h av, w3v;
        const size_t wb = ((size_t)e * HID + hc * HCHUNK + kb) * DIM + ncb + nl;
#pragma unroll
        for (int j = 0; j < 16; ++j) {
          const int k = (j < 8) ? (ks + j) : (ks + j + 8);
          av[j]  = acts[nl * HCHUNK + kb + k];
          w3v[j] = (_Float16)W3[wb + (size_t)k * DIM];
        }
        oacc[ot] = __builtin_amdgcn_wmma_f32_16x16x32_f16(false, av, false, w3v,
                                                          (short)0, oacc[ot],
                                                          false, false);
      }
    }
    __syncthreads();
  }
#pragma unroll 1
  for (int ot = 0; ot < 8; ++ot) {
    const int col = wid * 128 + ot * 16 + nl;
    const float b3v = b3[e * DIM + col];
#pragma unroll
    for (int r = 0; r < 8; ++r) {
      const int tok = ts[r + mo];
      if (tok >= 0)
        atomicAdd(&out[(size_t)tok * DIM + col], wt[r + mo] * (oacc[ot][r] + b3v));
    }
  }
}

// ---------------- host ----------------------------------------------------
extern "C" void kernel_launch(void* const* d_in, const int* in_sizes, int n_in,
                              void* d_out, int out_size, void* d_ws, size_t ws_size,
                              hipStream_t stream) {
  (void)in_sizes; (void)n_in; (void)out_size;
  const float* x  = (const float*)d_in[0];
  const float* Wg = (const float*)d_in[1];
  const float* W1 = (const float*)d_in[2];
  const float* b1 = (const float*)d_in[3];
  const float* W2 = (const float*)d_in[4];
  const float* b2 = (const float*)d_in[5];
  const float* W3 = (const float*)d_in[6];
  const float* b3 = (const float*)d_in[7];

  float* out = (float*)d_out;
  float* aux = out + AUX_OFF;
  float* gs  = out + GS_OFF;

  char*  ws     = (char*)d_ws;
  int*   counts = (int*)ws;
  int*   lists  = counts + NEXP;
  float* tokw   = (float*)(lists + NEXP * NTOK);
  float* lse2   = tokw + NTOK;
  _Float16* xh  = (_Float16*)(ws + XH_OFF);
  _Float16* W1t = (_Float16*)(ws + W1T_OFF);
  _Float16* W2t = (_Float16*)(ws + W2T_OFF);
  _Float16* W3t = (_Float16*)(ws + W3T_OFF);

  const bool pre = (ws_size >= WS_NEED);   // fixed per harness -> deterministic

  moe_zero<<<(OUT_N + 255) / 256, 256, 0, stream>>>(out, counts);
  moe_gate<<<NTOK / 8, 256, 0, stream>>>(x, Wg, gs, tokw, lse2, counts, lists);
  moe_aux<<<1, 256, 0, stream>>>(gs, lse2, aux);

  if (pre) {
    moe_cvt_x<<<(NTOK * DIM / 4 + 255) / 256, 256, 0, stream>>>(x, xh);
    const int nW = (int)(((size_t)NEXP * DIM * HID) / 256);
    moe_cvt_panels<DIM, HID><<<nW, 256, 0, stream>>>(W1, W1t);
    moe_cvt_panels<DIM, HID><<<nW, 256, 0, stream>>>(W2, W2t);
    moe_cvt_panels<HID, DIM><<<nW, 256, 0, stream>>>(W3, W3t);
    moe_expert32<<<dim3(NTOK / 32, NEXP), 256, 0, stream>>>(
        xh, (const v16h*)W1t, (const v16h*)W2t, (const v16h*)W3t,
        b1, b2, b3, counts, lists, tokw, out);
  } else {
    moe_expert16<<<dim3(NTOK / 16, NEXP), 256, 0, stream>>>(
        x, W1, W2, W3, b1, b2, b3, counts, lists, tokw, out);
  }
}
